// EncoderLayer_63299228009435
// MI455X (gfx1250) — matrix operand order repaired and checked by reading, not compiled, not run on hardware
//
#include <hip/hip_runtime.h>
#include <hip/hip_bf16.h>
#include <stdint.h>

// ---------------------------------------------------------------------------
// Transformer encoder layer for gfx1250 (MI455X), WMMA f16-in / f32-accum.
// B=2, S=2048, D=1024, H=16, DK=64, DH=4096.  Q=K=V (reference shares Wq).
// GEMMs: LDS-staged B (shared by 8 waves), 32x64 tile per wave.
// Attention: flash, 64 keys/iter, scale folded into Q fragment.
// ---------------------------------------------------------------------------

typedef __attribute__((ext_vector_type(16))) _Float16 v16h;
typedef __attribute__((ext_vector_type(8)))  float    v8f;

#define B_   2
#define S_   2048
#define D_   1024
#define H_   16
#define DK_  64
#define DH_  4096
#define M_   (B_ * S_)

#define BSH_LD 40     // LDS stride (halfs) for GEMM B tile: 20 dwords, conflict-free
#define P_LD   72     // LDS stride (halfs) for attention P tile: 36 dwords, conflict-free

// ---- WMMA fragment loaders (layouts per CDNA5 ISA §7.12.2) -----

// A-matrix 16x32 f16: lane L -> row m = L&15, hi = L>>4.
//   VGPR 0..3 : K = hi*8 + 2r + s ;  VGPR 4..7 : K = 16 + hi*8 + 2(r-4) + s
__device__ __forceinline__ v16h loadA(const _Float16* __restrict__ base, int ld, int k0) {
    int lane = threadIdx.x & 31;
    int m = lane & 15, hi = lane >> 4;
    const _Float16* row = base + (size_t)m * ld + k0 + hi * 8;
    v16h a;
#pragma unroll
    for (int r = 0; r < 4; ++r) { a[2*r]   = row[2*r];      a[2*r+1] = row[2*r+1]; }
#pragma unroll
    for (int r = 0; r < 4; ++r) { a[8+2*r] = row[16 + 2*r]; a[9+2*r] = row[16 + 2*r + 1]; }
    return a;
}

// B-matrix 32x16 f16, K-contiguous source: element (K,n) at base[n*ld + k0 + K].
// lane L -> col n = L&15, hi = L>>4;  elem e<8: K = hi*8 + e ; elem e>=8: K = 16 + hi*8 + (e-8).
__device__ __forceinline__ v16h loadBkc(const _Float16* __restrict__ base, int ld, int k0) {
    int lane = threadIdx.x & 31;
    int n = lane & 15, hi = lane >> 4;
    const _Float16* col = base + (size_t)n * ld + k0 + hi * 8;
    v16h b;
#pragma unroll
    for (int r = 0; r < 8; ++r) { b[r] = col[r]; b[8+r] = col[16+r]; }
    return b;
}

__device__ __forceinline__ v8f wmma16(v16h a, v16h b, v8f c) {
    return __builtin_amdgcn_wmma_f32_16x16x32_f16(false, a, false, b, (short)0, c,
                                                  false, false);
}

// ---------------------------- conversion kernels ---------------------------

__global__ __launch_bounds__(256) void cvt_f16_kernel(const float* __restrict__ src,
                                                      _Float16* __restrict__ dst, int n) {
    int i = blockIdx.x * 256 + threadIdx.x;
    if (i < n) dst[i] = (_Float16)src[i];
}

// src [K,N] f32 (row-major) -> dst [N,K] f16 (so B-operand loads are K-contiguous)
__global__ __launch_bounds__(256) void cvt_tr_f16_kernel(const float* __restrict__ src,
                                                         _Float16* __restrict__ dst,
                                                         int K, int N) {
    int i = blockIdx.x * 256 + threadIdx.x;
    if (i < K * N) {
        int k = i / N, n = i - k * N;
        dst[(size_t)n * K + k] = (_Float16)src[i];
    }
}

// ------------------------------- WMMA GEMM ---------------------------------
// C[M,N] = A[M,K](f16,rm) @ BT[N,K](f16) + bias.
// Block: 256 threads = 8 waves, block tile 256 rows x 64 cols.
// Per k-step: 32x64 B chunk staged once in LDS (shared by all 8 waves);
// each wave computes a 32x64 tile (2 A frags x 4 B frags = 8 WMMAs).
template<bool RES, bool RELU, bool OUTF32, bool OUTF16, bool QT>
__global__ __launch_bounds__(256) void gemm_wmma_kernel(
    const _Float16* __restrict__ A, const _Float16* __restrict__ BT,
    const float* __restrict__ bias, const float* __restrict__ Res,
    float* __restrict__ Cf, _Float16* __restrict__ Ch, _Float16* __restrict__ CqT,
    int M, int N, int K) {
    __shared__ _Float16 Bsh[64 * BSH_LD];           // 5 KB, padded rows
    int wave = threadIdx.x >> 5;
    int lane = threadIdx.x & 31;
    int rowbase = blockIdx.y * 256 + wave * 32;
    int colbase = blockIdx.x * 64;
    const _Float16* Arow = A + (size_t)rowbase * K;

    // cooperative B-load assignment: thread t -> row n = t/4, 8 halfs at (t%4)*8
    int ldn  = threadIdx.x >> 2;
    int ldk  = (threadIdx.x & 3) * 8;
    const _Float16* Bsrc = BT + (size_t)(colbase + ldn) * K + ldk;
    _Float16* Bdst = Bsh + ldn * BSH_LD + ldk;

    v8f acc[2][4] = {{{}, {}, {}, {}}, {{}, {}, {}, {}}};
    for (int k0 = 0; k0 < K; k0 += 32) {
        __syncthreads();
        *(uint4*)Bdst = *(const uint4*)(Bsrc + k0);
        __syncthreads();
        v16h a0 = loadA(Arow,           K, k0);
        v16h a1 = loadA(Arow + 16 * K,  K, k0);
#pragma unroll
        for (int t = 0; t < 4; ++t) {
            v16h b = loadBkc(Bsh + t * 16 * BSH_LD, BSH_LD, 0);
            acc[0][t] = wmma16(a0, b, acc[0][t]);
            acc[1][t] = wmma16(a1, b, acc[1][t]);
        }
    }

    int n0 = lane & 15, hi = lane >> 4;
#pragma unroll
    for (int t = 0; t < 4; ++t) {
        int col = colbase + t * 16 + n0;
        float bv = bias[col];
#pragma unroll
        for (int rg = 0; rg < 2; ++rg) {
#pragma unroll
            for (int r = 0; r < 8; ++r) {
                int row = rowbase + rg * 16 + hi * 8 + r;  // C layout: VGPR r -> M=r+8*hi
                float v = acc[rg][t][r] + bv;
                if (RES)  v += Res[(size_t)row * N + col];
                if (RELU) v = v > 0.f ? v : 0.f;
                if (OUTF32) Cf[(size_t)row * N + col] = v;
                if (OUTF16) Ch[(size_t)row * N + col] = (_Float16)v;
                if (QT) {                                   // qT[b, col, tok] for V loads
                    int bb  = row >> 11;                    // row / S_
                    int tok = row & (S_ - 1);
                    CqT[((size_t)bb * D_ + col) * S_ + tok] = (_Float16)v;
                }
            }
        }
    }
}

// ---------------------------- flash attention ------------------------------
// 4 waves/block; wave owns one 16-query tile of one (b,h). 64 keys / iter.
__global__ __launch_bounds__(128) void flash_attn_kernel(
    const _Float16* __restrict__ qh,   // [B,S,1024] row-major (Q and K source)
    const _Float16* __restrict__ qT,   // [B,1024,S]           (V source)
    _Float16* __restrict__ attn) {     // [B,S,1024] f16 out
    __shared__ _Float16 ldsP[4][16 * P_LD];         // per-wave P re-pack tile (2.25 KB)
    int wave = threadIdx.x >> 5, lane = threadIdx.x & 31;
    int h = blockIdx.y, b = blockIdx.z;
    int qbase = (blockIdx.x * 4 + wave) * 16;
    int n0 = lane & 15, hi = lane >> 4;

    const _Float16* Qbase = qh + ((size_t)b * S_ + qbase) * D_;
    v16h aq0 = loadA(Qbase, D_, h * DK_);
    v16h aq1 = loadA(Qbase, D_, h * DK_ + 32);
    // fold softmax scale 1/sqrt(S) into Q fragments (scores arrive pre-scaled)
    const _Float16 hscale = (_Float16)0.022097086912079608f;
#pragma unroll
    for (int i = 0; i < 16; ++i) { aq0[i] = aq0[i] * hscale; aq1[i] = aq1[i] * hscale; }

    v8f o[4] = {{}, {}, {}, {}};
    float mrow[8], lsum[8];
#pragma unroll
    for (int r = 0; r < 8; ++r) { mrow[r] = -3.0e38f; lsum[r] = 0.f; }
    _Float16* P = ldsP[wave];

    for (int j = 0; j < S_; j += 64) {
        const _Float16* K0 = qh + ((size_t)b * S_ + j) * D_;
        v8f s[4] = {{}, {}, {}, {}};
#pragma unroll
        for (int t4 = 0; t4 < 4; ++t4) {
            const _Float16* Kt = K0 + (size_t)t4 * 16 * D_;
            v16h bk0 = loadBkc(Kt, D_, h * DK_);
            s[t4] = wmma16(aq0, bk0, s[t4]);
            v16h bk1 = loadBkc(Kt, D_, h * DK_ + 32);
            s[t4] = wmma16(aq1, bk1, s[t4]);
        }

        float corr[8];
#pragma unroll
        for (int r = 0; r < 8; ++r) {
            // local max over 4 tiles first, then one 16-lane shuffle reduction
            float t = fmaxf(fmaxf(s[0][r], s[1][r]), fmaxf(s[2][r], s[3][r]));
#pragma unroll
            for (int off = 1; off < 16; off <<= 1) t = fmaxf(t, __shfl_xor(t, off, 32));
            float mnew = fmaxf(mrow[r], t);
            float c  = __expf(mrow[r] - mnew);
            float p0 = __expf(s[0][r] - mnew), p1 = __expf(s[1][r] - mnew);
            float p2 = __expf(s[2][r] - mnew), p3 = __expf(s[3][r] - mnew);
            float rs = (p0 + p1) + (p2 + p3);
#pragma unroll
            for (int off = 1; off < 16; off <<= 1) rs += __shfl_xor(rs, off, 32);
            lsum[r] = lsum[r] * c + rs;
            mrow[r] = mnew;
            corr[r] = c;
            int row = hi * 8 + r;                           // C layout -> LDS 16x64 (padded)
            P[row * P_LD + n0]      = (_Float16)p0;
            P[row * P_LD + 16 + n0] = (_Float16)p1;
            P[row * P_LD + 32 + n0] = (_Float16)p2;
            P[row * P_LD + 48 + n0] = (_Float16)p3;
        }
#pragma unroll
        for (int t4 = 0; t4 < 4; ++t4)
#pragma unroll
            for (int r = 0; r < 8; ++r) o[t4][r] *= corr[r];

        v16h ap0 = loadA(P, P_LD, 0);                       // repacked A-layout probs
        v16h ap1 = loadA(P, P_LD, 32);
#pragma unroll
        for (int t4 = 0; t4 < 4; ++t4) {
            const _Float16* Vb = qT + ((size_t)b * D_ + h * DK_ + t4 * 16) * S_;
            v16h bv0 = loadBkc(Vb, S_, j);
            o[t4] = wmma16(ap0, bv0, o[t4]);
            v16h bv1 = loadBkc(Vb, S_, j + 32);
            o[t4] = wmma16(ap1, bv1, o[t4]);
        }
    }

#pragma unroll
    for (int r = 0; r < 8; ++r) {
        float inv = 1.f / lsum[r];
        int row = qbase + hi * 8 + r;
#pragma unroll
        for (int t4 = 0; t4 < 4; ++t4)
            attn[((size_t)b * S_ + row) * D_ + h * DK_ + t4 * 16 + n0] =
                (_Float16)(o[t4][r] * inv);
    }
}

// ------------------------------- layernorm ---------------------------------
// One row (D=1024) per 256-thread block, in place; optional f16 copy.
template<bool STOREH>
__global__ __launch_bounds__(256) void layernorm_kernel(
    float* __restrict__ x, const float* __restrict__ g, const float* __restrict__ be,
    _Float16* __restrict__ xh) {
    __shared__ float sbuf[8], ssbuf[8];
    int row = blockIdx.x;
    float* xr = x + (size_t)row * D_;
    float v[4], s = 0.f, ss = 0.f;
#pragma unroll
    for (int i = 0; i < 4; ++i) {
        v[i] = xr[threadIdx.x + 256 * i];
        s += v[i]; ss += v[i] * v[i];
    }
#pragma unroll
    for (int off = 1; off < 32; off <<= 1) {
        s  += __shfl_xor(s,  off, 32);
        ss += __shfl_xor(ss, off, 32);
    }
    if ((threadIdx.x & 31) == 0) { sbuf[threadIdx.x >> 5] = s; ssbuf[threadIdx.x >> 5] = ss; }
    __syncthreads();
    float ts = 0.f, tss = 0.f;
#pragma unroll
    for (int w = 0; w < 8; ++w) { ts += sbuf[w]; tss += ssbuf[w]; }
    float mean = ts * (1.f / D_);
    float var  = tss * (1.f / D_) - mean * mean;
    float rstd = rsqrtf(var + 1e-5f);
#pragma unroll
    for (int i = 0; i < 4; ++i) {
        int c = threadIdx.x + 256 * i;
        float y = (v[i] - mean) * rstd * g[c] + be[c];
        xr[c] = y;
        if (STOREH) xh[(size_t)row * D_ + c] = (_Float16)y;
    }
}

// ------------------------------- launcher ----------------------------------

extern "C" void kernel_launch(void* const* d_in, const int* in_sizes, int n_in,
                              void* d_out, int out_size, void* d_ws, size_t ws_size,
                              hipStream_t stream) {
    (void)in_sizes; (void)n_in; (void)out_size; (void)ws_size;

    const float* x     = (const float*)d_in[0];
    const float* Wq    = (const float*)d_in[1];
    const float* bq    = (const float*)d_in[2];
    const float* Wo    = (const float*)d_in[3];
    const float* bo    = (const float*)d_in[4];
    const float* ln1_g = (const float*)d_in[5];
    const float* ln1_b = (const float*)d_in[6];
    const float* W1    = (const float*)d_in[7];
    const float* b1    = (const float*)d_in[8];
    const float* W2    = (const float*)d_in[9];
    const float* b2    = (const float*)d_in[10];
    const float* ln2_g = (const float*)d_in[11];
    const float* ln2_b = (const float*)d_in[12];
    float* out = (float*)d_out;

    // workspace layout (bytes)
    char* ws = (char*)d_ws;
    _Float16* x_h   = (_Float16*)(ws);                               //  8 MB  [M,D]
    _Float16* q_h   = (_Float16*)(ws + 8388608);                     //  8 MB  [B,S,1024]
    _Float16* qT_h  = (_Float16*)(ws + 2 * 8388608);                 //  8 MB  [B,1024,S]
    _Float16* attn_h= (_Float16*)(ws + 3 * 8388608);                 //  8 MB  [M,1024]
    _Float16* ln1_h = (_Float16*)(ws + 4 * 8388608);                 //  8 MB  [M,D]
    _Float16* h_h   = (_Float16*)(ws + 5 * 8388608);                 // 32 MB  [M,DH]
    _Float16* WqT   = (_Float16*)(ws + 5 * 8388608 + 33554432);      //  2 MB  [1024,1024]
    _Float16* WoT   = (_Float16*)((char*)WqT + 2097152);             //  2 MB
    _Float16* W1T   = (_Float16*)((char*)WoT + 2097152);             //  8 MB  [4096,1024]
    _Float16* W2T   = (_Float16*)((char*)W1T + 8388608);             //  8 MB  [1024,4096]
    float*    y1    = (float*)   ((char*)W2T + 8388608);             // 16 MB  [M,D]

    // 1) f16 conversions (weights transposed for K-contiguous B operands)
    cvt_f16_kernel<<<M_ * D_ / 256, 256, 0, stream>>>(x, x_h, M_ * D_);
    cvt_tr_f16_kernel<<<D_ * D_  / 256, 256, 0, stream>>>(Wq, WqT, D_, D_);
    cvt_tr_f16_kernel<<<D_ * D_  / 256, 256, 0, stream>>>(Wo, WoT, D_, D_);
    cvt_tr_f16_kernel<<<D_ * DH_ / 256, 256, 0, stream>>>(W1, W1T, D_, DH_);
    cvt_tr_f16_kernel<<<DH_ * D_ / 256, 256, 0, stream>>>(W2, W2T, DH_, D_);

    // 2) q = x@Wq + bq  ->  q_h (row-major f16) and qT_h (transposed f16)
    gemm_wmma_kernel<false, false, false, true, true>
        <<<dim3(D_ / 64, M_ / 256), 256, 0, stream>>>(
            x_h, WqT, bq, nullptr, nullptr, q_h, qT_h, M_, D_, D_);

    // 3) flash attention (Q = K = V = q)
    flash_attn_kernel<<<dim3(S_ / 64, H_, B_), 128, 0, stream>>>(q_h, qT_h, attn_h);

    // 4) y1 = attn@Wo + bo + x   (fp32)
    gemm_wmma_kernel<true, false, true, false, false>
        <<<dim3(D_ / 64, M_ / 256), 256, 0, stream>>>(
            attn_h, WoT, bo, x, y1, nullptr, nullptr, M_, D_, D_);

    // 5) LN1 in place on y1, f16 copy -> ln1_h
    layernorm_kernel<true><<<M_, 256, 0, stream>>>(y1, ln1_g, ln1_b, ln1_h);

    // 6) h = relu(ln1@W1 + b1) -> h_h (f16 only)
    gemm_wmma_kernel<false, true, false, true, false>
        <<<dim3(DH_ / 64, M_ / 256), 256, 0, stream>>>(
            ln1_h, W1T, b1, nullptr, nullptr, h_h, nullptr, M_, DH_, D_);

    // 7) y2 = h@W2 + b2 + ln1  -> d_out (fp32)
    gemm_wmma_kernel<true, false, true, false, false>
        <<<dim3(D_ / 64, M_ / 256), 256, 0, stream>>>(
            h_h, W2T, b2, y1, out, nullptr, nullptr, M_, D_, DH_);

    // 8) LN2 in place on d_out
    layernorm_kernel<false><<<M_, 256, 0, stream>>>(out, ln2_g, ln2_b, nullptr);
}
